// Spatial_Attn_75917841924503
// MI455X (gfx1250) — matrix operand order repaired and checked by reading, not compiled, not run on hardware
//
#include <hip/hip_runtime.h>

// ---------------------------------------------------------------------------
// Spatial self-attention (SAGAN), fused flash-attention for MI455X (gfx1250).
// B=4, C=256, N=4096, dk=32. bf16 WMMA (v_wmma_f32_16x16x32_bf16), f32 accum.
// K/V blocks (64 keys) staged to LDS with global_load_async_to_lds_b128
// (ASYNCcnt), double-buffered; next block's DMA is issued BEFORE draining the
// current one (s_wait_asynccnt <= 9) so the async engine runs ahead of WMMA.
// ---------------------------------------------------------------------------

#define BB   4
#define CCH  256
#define NSP  4096      // 64*64 spatial
#define DKQ  32
#define KBLK 64        // keys per flash block

typedef __attribute__((ext_vector_type(16))) __bf16        v16bf;
typedef __attribute__((ext_vector_type(8)))  float         v8f;
typedef __attribute__((ext_vector_type(8)))  unsigned int  v8u;

union BFPack { v8u u; v16bf b; };

// round-to-nearest-even f32 -> bf16
__device__ __forceinline__ unsigned short f2bf(float f) {
  unsigned u = __float_as_uint(f);
  unsigned r = u + 0x7FFFu + ((u >> 16) & 1u);
  return (unsigned short)(r >> 16);
}

// Load 16 bf16 (two 16B chunks) from a row into a WMMA operand register set.
__device__ __forceinline__ v16bf load_ab(const unsigned short* row, int off0, int off1) {
  uint4 lo = *reinterpret_cast<const uint4*>(row + off0);
  uint4 hi = *reinterpret_cast<const uint4*>(row + off1);
  BFPack p;
  p.u = (v8u){lo.x, lo.y, lo.z, lo.w, hi.x, hi.y, hi.z, hi.w};
  return p.b;
}

__device__ __forceinline__ v8f wmma_bf16(v16bf a, v16bf b, v8f c) {
  // D = A(16x32 bf16) * B(32x16 bf16) + C(16x16 f32)
  return __builtin_amdgcn_wmma_f32_16x16x32_bf16(false, a, false, b, (short)0, c,
                                                 false, false);
}

// ---- CDNA5 async copy global -> LDS (16 bytes per lane, ASYNCcnt) ----------
__device__ __forceinline__ void async_copy16(unsigned lds_off,
                                             const unsigned short* gptr) {
  asm volatile("global_load_async_to_lds_b128 %0, %1, off"
               :: "v"(lds_off), "v"(gptr)
               : "memory");
}

__device__ __forceinline__ void wait_async0() {
#if __has_builtin(__builtin_amdgcn_s_wait_asynccnt)
  __builtin_amdgcn_s_wait_asynccnt(0);
#else
  asm volatile("s_wait_asynccnt 0x0" ::: "memory");
#endif
}
// Drain everything except the 9 async ops just issued for the next block.
__device__ __forceinline__ void wait_async9() {
#if __has_builtin(__builtin_amdgcn_s_wait_asynccnt)
  __builtin_amdgcn_s_wait_asynccnt(9);
#else
  asm volatile("s_wait_asynccnt 0x9" ::: "memory");
#endif
}

// ds_swizzle xor-reductions within each 16-lane half (row of a WMMA C tile).
#define SWZ(v, mask) \
  __int_as_float(__builtin_amdgcn_ds_swizzle(__float_as_int(v), (((mask) << 10) | 0x1f)))

__device__ __forceinline__ float hmax16(float v) {
  v = fmaxf(v, SWZ(v, 1));
  v = fmaxf(v, SWZ(v, 2));
  v = fmaxf(v, SWZ(v, 4));
  v = fmaxf(v, SWZ(v, 8));
  return v;
}
__device__ __forceinline__ float hsum16(float v) {
  v += SWZ(v, 1);
  v += SWZ(v, 2);
  v += SWZ(v, 4);
  v += SWZ(v, 8);
  return v;
}

// ---------------------------------------------------------------------------
// f32 -> bf16 convert (weights)
// ---------------------------------------------------------------------------
__global__ void cvt_bf16_kernel(const float* __restrict__ in,
                                unsigned short* __restrict__ out, int n) {
  int i = blockIdx.x * blockDim.x + threadIdx.x;
  if (i < n) out[i] = f2bf(in[i]);
}

// ---------------------------------------------------------------------------
// x[b][c][n] f32 -> xt[b][n][c] bf16   (K-dim contiguous for WMMA B loads)
// ---------------------------------------------------------------------------
__global__ void xpose_kernel(const float* __restrict__ x,
                             unsigned short* __restrict__ xt) {
  int i = blockIdx.x * 256 + threadIdx.x;      // 0 .. B*C*N-1, n fastest
  int n = i & (NSP - 1);
  int c = (i >> 12) & (CCH - 1);
  int b = i >> 20;
  xt[((size_t)(b * NSP + n)) * CCH + c] = f2bf(x[i]);
}

// ---------------------------------------------------------------------------
// Projection GEMM:  Y[m][n] = sum_c W[m][c] * x[c][n]  (+ bias[m])
//   A = Wbf (M x 256, row-major bf16), B(c,n) = xt[b][n][c].
//   mode 0: out[b][m][n]  (V, channel-major)   mode 1: out[b][n][m]  (Q/K)
// One 16x16 output tile per wave, 8 waves per block.
// ---------------------------------------------------------------------------
__global__ void __launch_bounds__(256) proj_kernel(
    const unsigned short* __restrict__ Wbf, const float* __restrict__ bias,
    const unsigned short* __restrict__ xt, unsigned short* __restrict__ out,
    int M, int mode) {
  const int lane = threadIdx.x & 31;
  const int wave = threadIdx.x >> 5;
  const int b = blockIdx.y;
  const int tile = blockIdx.x * 8 + wave;
  const int mt = tile / (NSP / 16);
  const int nt = tile % (NSP / 16);
  const int m16 = lane & 15;
  const int h = lane >> 4;

  const unsigned short* arow = Wbf + (size_t)(mt * 16 + m16) * 256;
  const unsigned short* brow = xt + ((size_t)b * NSP + nt * 16 + m16) * 256;

  v8f acc;
#pragma unroll
  for (int r = 0; r < 8; ++r) acc[r] = 0.0f;

#pragma unroll
  for (int kk = 0; kk < 256; kk += 32) {
    v16bf a = load_ab(arow, kk + 8 * h, kk + 16 + 8 * h);
    v16bf bm = load_ab(brow, kk + 8 * h, kk + 16 + 8 * h);
    acc = wmma_bf16(a, bm, acc);
  }

#pragma unroll
  for (int r = 0; r < 8; ++r) {
    int mg = mt * 16 + r + 8 * h;
    int ng = nt * 16 + m16;
    unsigned short bvv = f2bf(acc[r] + bias[mg]);
    if (mode == 0)
      out[((size_t)(b * M + mg)) * NSP + ng] = bvv;       // V: [c][n]
    else
      out[((size_t)b * NSP + ng) * M + mg] = bvv;         // Q/K: [n][dk]
  }
}

// ---------------------------------------------------------------------------
// Fused flash attention + gamma residual.
//   qw/kw: [B][N][32] bf16, vt: [B][256][N] bf16, x/out: [B][256][N] f32.
// Workgroup: 256 threads = 8 waves; wave owns 16 query rows, full 256-ch out.
// K/V key-blocks (64 keys) staged to LDS asynchronously, double-buffered.
// Every wave issues exactly 9 async ops per stage (1 K + 8 V chunks/lane).
// ---------------------------------------------------------------------------
__global__ void __launch_bounds__(256) flash_kernel(
    const unsigned short* __restrict__ qw, const unsigned short* __restrict__ kw,
    const unsigned short* __restrict__ vt, const float* __restrict__ x,
    const float* __restrict__ gamma, float* __restrict__ out) {
  __shared__ __align__(16) unsigned short klds[2][KBLK][DKQ];   //  8 KB
  __shared__ __align__(16) unsigned short vlds[2][CCH][KBLK];   // 64 KB
  __shared__ __align__(16) unsigned short plds[8][16][KBLK];    // 16 KB

  const int tid = threadIdx.x;
  const int lane = tid & 31;
  const int wave = tid >> 5;
  const int b = blockIdx.y;
  const int q0 = blockIdx.x * 128 + wave * 16;
  const int m16 = lane & 15;
  const int h = lane >> 4;

  const unsigned kbase = (unsigned)(uintptr_t)&klds[0][0][0];
  const unsigned vbase = (unsigned)(uintptr_t)&vlds[0][0][0];

  // ---- async stage of one 64-key K/V block into LDS buffer `buf` ----
  auto stage_kv = [&](int buf, int kb) {
    // K tile: 64 rows x 64B = 4 KB contiguous; one 16B chunk per thread
    async_copy16(kbase + buf * 4096 + tid * 16,
                 kw + ((size_t)b * NSP + kb) * DKQ + tid * 8);
    // V tile: 256 channel rows x 128B; thread t stages channel t's row
    const unsigned short* vg = vt + ((size_t)(b * CCH + tid)) * NSP + kb;
    unsigned voff = vbase + buf * 32768 + tid * 128;
#pragma unroll
    for (int j = 0; j < 8; ++j) async_copy16(voff + j * 16, vg + j * 8);
  };

  // Q tile in A-matrix layout (16 rows x dk=32)
  const unsigned short* qrow = qw + ((size_t)b * NSP + q0 + m16) * DKQ;
  const v16bf qa = load_ab(qrow, 8 * h, 16 + 8 * h);

  v8f o[16];                 // 16x256 f32 output accumulator (C-layout tiles)
  float m_run[8], l_run[8];  // per-row running max / sum (rows r + 8*h)
#pragma unroll
  for (int ct = 0; ct < 16; ++ct)
#pragma unroll
    for (int r = 0; r < 8; ++r) o[ct][r] = 0.0f;
#pragma unroll
  for (int r = 0; r < 8; ++r) { m_run[r] = -3.0e38f; l_run[r] = 0.0f; }

  stage_kv(0, 0);  // prologue: fill buffer 0
  int buf = 0;

  for (int kb = 0; kb < NSP; kb += KBLK) {
    // Kick off next block's DMA first, then drain this block's 9 older ops.
    if (kb + KBLK < NSP) {
      stage_kv(buf ^ 1, kb + KBLK);
      wait_async9();
    } else {
      wait_async0();
    }
    __syncthreads();  // this block's K/V visible to all waves

    // ---- S = Q * K^T : 16 x 64, four 16x16 WMMA tiles, K-dim = dk = 32 ----
    v8f s[4];
#pragma unroll
    for (int t = 0; t < 4; ++t) {
      v16bf kbm = load_ab(&klds[buf][t * 16 + m16][0], 8 * h, 16 + 8 * h);
      v8f z;
#pragma unroll
      for (int r = 0; r < 8; ++r) z[r] = 0.0f;
      s[t] = wmma_bf16(qa, kbm, z);
    }

    // ---- online softmax over this key block ----
#pragma unroll
    for (int r = 0; r < 8; ++r) {
      float mx = fmaxf(fmaxf(s[0][r], s[1][r]), fmaxf(s[2][r], s[3][r]));
      mx = hmax16(mx);
      float mn = fmaxf(m_run[r], mx);
      float corr = __expf(m_run[r] - mn);
      m_run[r] = mn;
      float rs = 0.0f;
#pragma unroll
      for (int t = 0; t < 4; ++t) {
        float p = __expf(s[t][r] - mn);
        s[t][r] = p;
        rs += p;
      }
      rs = hsum16(rs);
      l_run[r] = l_run[r] * corr + rs;
#pragma unroll
      for (int ct = 0; ct < 16; ++ct) o[ct][r] *= corr;  // rescale accum
    }

    // ---- P: C-layout -> A-layout via wave-private LDS round-trip ----
#pragma unroll
    for (int t = 0; t < 4; ++t)
#pragma unroll
      for (int r = 0; r < 8; ++r)
        plds[wave][r + 8 * h][t * 16 + m16] = f2bf(s[t][r]);
    // same-wave DS ops are in-order; compiler inserts s_wait_dscnt
    v16bf pa0 = load_ab(&plds[wave][m16][0], 8 * h, 16 + 8 * h);   // keys 0..31
    v16bf pa1 = load_ab(&plds[wave][m16][32], 8 * h, 16 + 8 * h);  // keys 32..63

    // ---- O += P * V : 16 x 256, K-dim = 64 keys = 2 WMMAs per chan tile ----
#pragma unroll
    for (int ct = 0; ct < 16; ++ct) {
      const unsigned short* vrow = &vlds[buf][ct * 16 + m16][0];
      v16bf b0 = load_ab(vrow, 8 * h, 16 + 8 * h);
      v16bf b1 = load_ab(vrow + 32, 8 * h, 16 + 8 * h);
      o[ct] = wmma_bf16(pa0, b0, o[ct]);
      o[ct] = wmma_bf16(pa1, b1, o[ct]);
    }

    __syncthreads();  // all waves done with `buf` before it is re-staged
    buf ^= 1;
  }

  // ---- epilogue: out = gamma * (O / l) + x ----
  const float g = gamma[0];
  float inv[8];
#pragma unroll
  for (int r = 0; r < 8; ++r) inv[r] = g / l_run[r];
#pragma unroll
  for (int ct = 0; ct < 16; ++ct) {
    int ch = ct * 16 + m16;
#pragma unroll
    for (int r = 0; r < 8; ++r) {
      size_t idx = ((size_t)(b * CCH + ch)) * NSP + (q0 + r + 8 * h);
      out[idx] = o[ct][r] * inv[r] + x[idx];
    }
  }
}

// ---------------------------------------------------------------------------
extern "C" void kernel_launch(void* const* d_in, const int* in_sizes, int n_in,
                              void* d_out, int out_size, void* d_ws, size_t ws_size,
                              hipStream_t stream) {
  const float* x     = (const float*)d_in[0];
  const float* Wq    = (const float*)d_in[1];
  const float* bq    = (const float*)d_in[2];
  const float* Wk    = (const float*)d_in[3];
  const float* bk    = (const float*)d_in[4];
  const float* Wv    = (const float*)d_in[5];
  const float* bv    = (const float*)d_in[6];
  const float* gamma = (const float*)d_in[7];
  float* out = (float*)d_out;

  unsigned char* ws = (unsigned char*)d_ws;
  unsigned short* xt  = (unsigned short*)ws; ws += (size_t)BB * NSP * CCH * 2;  // 8 MB
  unsigned short* wqb = (unsigned short*)ws; ws += (size_t)DKQ * CCH * 2;
  unsigned short* wkb = (unsigned short*)ws; ws += (size_t)DKQ * CCH * 2;
  unsigned short* wvb = (unsigned short*)ws; ws += (size_t)CCH * CCH * 2;
  unsigned short* qws = (unsigned short*)ws; ws += (size_t)BB * NSP * DKQ * 2;  // 1 MB
  unsigned short* kws = (unsigned short*)ws; ws += (size_t)BB * NSP * DKQ * 2;  // 1 MB
  unsigned short* vtw = (unsigned short*)ws;                                    // 8 MB

  // 1) precision conversion + transpose
  cvt_bf16_kernel<<<(DKQ * CCH + 255) / 256, 256, 0, stream>>>(Wq, wqb, DKQ * CCH);
  cvt_bf16_kernel<<<(DKQ * CCH + 255) / 256, 256, 0, stream>>>(Wk, wkb, DKQ * CCH);
  cvt_bf16_kernel<<<(CCH * CCH + 255) / 256, 256, 0, stream>>>(Wv, wvb, CCH * CCH);
  xpose_kernel<<<(BB * CCH * NSP) / 256, 256, 0, stream>>>(x, xt);

  // 2) Q/K/V projections (WMMA GEMMs)
  //    Q,K: M=32 -> 512 tiles/batch -> grid.x = 64 ;  V: M=256 -> 4096 tiles
  proj_kernel<<<dim3(64, BB), 256, 0, stream>>>(wqb, bq, xt, qws, DKQ, 1);
  proj_kernel<<<dim3(64, BB), 256, 0, stream>>>(wkb, bk, xt, kws, DKQ, 1);
  proj_kernel<<<dim3(512, BB), 256, 0, stream>>>(wvb, bv, xt, vtw, CCH, 0);

  // 3) fused flash attention + residual: 32 query-blocks x 4 batches
  flash_kernel<<<dim3(NSP / 128, BB), 256, 0, stream>>>(qws, kws, vtw, x, gamma, out);
}
